// EncoderFLAX_60404420051391
// MI455X (gfx1250) — matrix operand order repaired and checked by reading, not compiled, not run on hardware
//
#include <hip/hip_runtime.h>
#include <hip/hip_bf16.h>
#include <math.h>

// Bidirectional LSTM encoder for MI455X (gfx1250).
//   1) transpose_to_bf16 : Wx/Wh (K x 4H fp32) -> (4H x K bf16), LDS-tiled.
//   2) embed_gather      : gather + fp32->bf16, time-major X[s][b][e].
//   3) gemm_xw (WMMA)    : Gpre = X @ WxT + bias. A and B double-buffered as
//                          64-wide K slabs via the async engine (ASYNCcnt);
//                          inner loop = DS reads + WMMA only. ~78 KB LDS ->
//                          4 blocks/WGP for cross-wave latency hiding.
//   4) lstm_step (WMMA)  : per step both dirs; K split across 4 waves, h tile
//                          async-staged, A/B frags register-pipelined, LDS
//                          reduction + fused gate math. One-shot streams use
//                          TH_NT so WhT stays L2-resident across the 512 steps.

typedef __attribute__((ext_vector_type(16))) __bf16 v16bf;
typedef __attribute__((ext_vector_type(8)))  __bf16 v8bf;
typedef __attribute__((ext_vector_type(8)))  float  v8f;
typedef __attribute__((ext_vector_type(8)))  unsigned short ushort8;

#define S_LEN 512
#define BATCH 64
#define KDIM  1024   // E == H == 1024
#define G4    4096   // 4*H
#define LROW  1032   // padded LDS row (shorts) for lstm_step h tile
#define AROW  68     // padded LDS row (shorts) for 64-wide A slabs (bank step 34)
#define BROW  68     // padded LDS row (shorts) for 64-wide B slabs

__device__ __forceinline__ unsigned short f2bf(float f) {
  unsigned int u = __float_as_uint(f);
  u += 0x7FFFu + ((u >> 16) & 1u);          // round-to-nearest-even
  return (unsigned short)(u >> 16);
}
__device__ __forceinline__ float sigm(float x) { return 1.0f / (1.0f + __expf(-x)); }

// CDNA5 async copy: global -> LDS, 16B per lane, tracked by ASYNCcnt.
__device__ __forceinline__ void async_copy_b128(const unsigned short* lds_dst,
                                                const unsigned short* gsrc) {
  unsigned lds_off = (unsigned)(unsigned long long)(uintptr_t)lds_dst;
  asm volatile("global_load_async_to_lds_b128 %0, %1, off"
               :: "v"(lds_off), "v"(gsrc) : "memory");
}
__device__ __forceinline__ void wait_async0() {
  asm volatile("s_wait_asynccnt 0x0" ::: "memory");
}

// A fragment (16x32 bf16, M x K): lane L<16 -> row M=L, K 0..15; L>=16 -> K 16..31
__device__ __forceinline__ v16bf load_frag_a(const unsigned short* row, int k0, int kh) {
  v8bf lo = *(const v8bf*)(row + k0 + kh * 16);
  v8bf hi = *(const v8bf*)(row + k0 + kh * 16 + 8);
  return __builtin_shufflevector(lo, hi, 0,1,2,3,4,5,6,7,8,9,10,11,12,13,14,15);
}
// B fragment (32x16 bf16, K x N) from BT stored N x K row-major (any row stride)
__device__ __forceinline__ v16bf load_frag_b_s(const unsigned short* p) {
  v8bf lo = *(const v8bf*)(p);
  v8bf hi = *(const v8bf*)(p + 8);
  return __builtin_shufflevector(lo, hi, 0,1,2,3,4,5,6,7,8,9,10,11,12,13,14,15);
}

// ---------------- 1) weight transpose + bf16 convert (LDS tiled) ----------------
__global__ void transpose_to_bf16(const float* __restrict__ W0, const float* __restrict__ W1,
                                  const float* __restrict__ W2, const float* __restrict__ W3,
                                  unsigned short* __restrict__ T0, unsigned short* __restrict__ T1,
                                  unsigned short* __restrict__ T2, unsigned short* __restrict__ T3) {
  __shared__ float tile[32][33];
  const float* W; unsigned short* T;
  switch (blockIdx.z) {
    case 0:  W = W0; T = T0; break;
    case 1:  W = W1; T = T1; break;
    case 2:  W = W2; T = T2; break;
    default: W = W3; T = T3; break;
  }
  const int n0 = blockIdx.x * 32;
  const int k0 = blockIdx.y * 32;
  const int tx = threadIdx.x, ty = threadIdx.y;   // 32 x 8
#pragma unroll
  for (int i = 0; i < 32; i += 8)
    tile[ty + i][tx] = W[(size_t)(k0 + ty + i) * G4 + n0 + tx];
  __syncthreads();
#pragma unroll
  for (int i = 0; i < 32; i += 8)
    T[(size_t)(n0 + ty + i) * KDIM + k0 + tx] = f2bf(tile[tx][ty + i]);
}

// ---------------- 2) embedding gather, fp32 -> bf16, time-major ----------------
__global__ void embed_gather(const int* __restrict__ idx, const float* __restrict__ tbl,
                             unsigned short* __restrict__ X) {
  size_t t = (size_t)blockIdx.x * blockDim.x + threadIdx.x;
  int e8 = (int)(t & 127);
  size_t bs = t >> 7;
  int s = (int)(bs & (S_LEN - 1));
  int b = (int)(bs >> 9);
  int row = idx[(size_t)b * S_LEN + s];
  const float* p = tbl + (size_t)row * KDIM + e8 * 8;
  ushort8 o;
#pragma unroll
  for (int i = 0; i < 8; ++i) o[i] = f2bf(p[i]);
  __builtin_nontemporal_store(o, (ushort8*)(X + ((size_t)s * BATCH + b) * KDIM + e8 * 8));
}

// slab stagers: 64-wide K slabs, 16B chunks spread over 128 threads
__device__ __forceinline__ void stage_a_slab(unsigned short* dst, const unsigned short* src,
                                             int k0, int tid) {
#pragma unroll
  for (int rnd = 0; rnd < 2; ++rnd) {          // 32 rows x 8 chunks
    int chunk = rnd * 128 + tid;
    int row = chunk >> 3, sub = (chunk & 7) * 8;
    async_copy_b128(dst + row * AROW + sub, src + (size_t)row * KDIM + k0 + sub);
  }
}
__device__ __forceinline__ void stage_b_slab(unsigned short* dst, const unsigned short* src,
                                             int k0, int tid) {
  const int brow0 = tid >> 3, bsub = (tid & 7) * 8;
#pragma unroll 4
  for (int rnd = 0; rnd < 16; ++rnd) {         // 256 rows x 8 chunks
    int row = rnd * 16 + brow0;
    async_copy_b128(dst + row * BROW + bsub, src + (size_t)row * KDIM + k0 + bsub);
  }
}

// ---------------- 3) batched input projection: Gpre = X @ WxT + bias ----------------
// grid (1024, 16, 2), block 128 (4 waves). Block tile 32(M) x 256(N).
__global__ void gemm_xw(const unsigned short* __restrict__ X,
                        const unsigned short* __restrict__ WxTf, const unsigned short* __restrict__ WxTb,
                        const float* __restrict__ bf_, const float* __restrict__ bb_,
                        float* __restrict__ Gf, float* __restrict__ Gb) {
  __shared__ __align__(16) unsigned short As[2][32 * AROW];    // 2 x 4.25 KB
  __shared__ __align__(16) unsigned short Bt[2][256 * BROW];   // 2 x 34 KB
  const int dir = blockIdx.z;
  const unsigned short* BT = dir ? WxTb : WxTf;
  const float* bias = dir ? bb_ : bf_;
  float* G = dir ? Gb : Gf;

  const int tid  = threadIdx.x;
  const int lane = tid & 31;
  const int wave = tid >> 5;
  const int mtile = blockIdx.x * 32;
  const int nbase = blockIdx.y * 256;
  const int r = lane & 15, kh = lane >> 4;

  const unsigned short* Xa  = X + (size_t)mtile * KDIM;
  const unsigned short* BTb = BT + (size_t)nbase * KDIM;

  stage_a_slab(As[0], Xa, 0, tid);
  stage_b_slab(Bt[0], BTb, 0, tid);
  wait_async0();
  __syncthreads();

  const int jstride = 16 * BROW;
  v8f acc0[4] = {}, acc1[4] = {};
  int cur = 0;
  for (int k0 = 0; k0 < KDIM; k0 += 64) {
    if (k0 + 64 < KDIM) {                      // async-prefetch next slabs
      stage_a_slab(As[cur ^ 1], Xa, k0 + 64, tid);
      stage_b_slab(Bt[cur ^ 1], BTb, k0 + 64, tid);
    }
    const unsigned short* ar0 = As[cur] + r * AROW;
    const unsigned short* ar1 = As[cur] + (16 + r) * AROW;
    const unsigned short* bsw = Bt[cur] + (size_t)(wave * 64 + r) * BROW + kh * 16;

    v16bf a00 = load_frag_a(ar0, 0, kh),  a10 = load_frag_a(ar1, 0, kh);
    v16bf a01 = load_frag_a(ar0, 32, kh), a11 = load_frag_a(ar1, 32, kh);
    v16bf b00 = load_frag_b_s(bsw + 0 * jstride);
    v16bf b01 = load_frag_b_s(bsw + 1 * jstride);
    v16bf b02 = load_frag_b_s(bsw + 2 * jstride);
    v16bf b03 = load_frag_b_s(bsw + 3 * jstride);
    v16bf b10 = load_frag_b_s(bsw + 0 * jstride + 32);
    v16bf b11 = load_frag_b_s(bsw + 1 * jstride + 32);
    v16bf b12 = load_frag_b_s(bsw + 2 * jstride + 32);
    v16bf b13 = load_frag_b_s(bsw + 3 * jstride + 32);

    acc0[0] = __builtin_amdgcn_wmma_f32_16x16x32_bf16(false, a00, false, b00, (short)0, acc0[0], false, false);
    acc1[0] = __builtin_amdgcn_wmma_f32_16x16x32_bf16(false, a10, false, b00, (short)0, acc1[0], false, false);
    acc0[1] = __builtin_amdgcn_wmma_f32_16x16x32_bf16(false, a00, false, b01, (short)0, acc0[1], false, false);
    acc1[1] = __builtin_amdgcn_wmma_f32_16x16x32_bf16(false, a10, false, b01, (short)0, acc1[1], false, false);
    acc0[2] = __builtin_amdgcn_wmma_f32_16x16x32_bf16(false, a00, false, b02, (short)0, acc0[2], false, false);
    acc1[2] = __builtin_amdgcn_wmma_f32_16x16x32_bf16(false, a10, false, b02, (short)0, acc1[2], false, false);
    acc0[3] = __builtin_amdgcn_wmma_f32_16x16x32_bf16(false, a00, false, b03, (short)0, acc0[3], false, false);
    acc1[3] = __builtin_amdgcn_wmma_f32_16x16x32_bf16(false, a10, false, b03, (short)0, acc1[3], false, false);
    acc0[0] = __builtin_amdgcn_wmma_f32_16x16x32_bf16(false, a01, false, b10, (short)0, acc0[0], false, false);
    acc1[0] = __builtin_amdgcn_wmma_f32_16x16x32_bf16(false, a11, false, b10, (short)0, acc1[0], false, false);
    acc0[1] = __builtin_amdgcn_wmma_f32_16x16x32_bf16(false, a01, false, b11, (short)0, acc0[1], false, false);
    acc1[1] = __builtin_amdgcn_wmma_f32_16x16x32_bf16(false, a11, false, b11, (short)0, acc1[1], false, false);
    acc0[2] = __builtin_amdgcn_wmma_f32_16x16x32_bf16(false, a01, false, b12, (short)0, acc0[2], false, false);
    acc1[2] = __builtin_amdgcn_wmma_f32_16x16x32_bf16(false, a11, false, b12, (short)0, acc1[2], false, false);
    acc0[3] = __builtin_amdgcn_wmma_f32_16x16x32_bf16(false, a01, false, b13, (short)0, acc0[3], false, false);
    acc1[3] = __builtin_amdgcn_wmma_f32_16x16x32_bf16(false, a11, false, b13, (short)0, acc1[3], false, false);

    wait_async0();                  // next slabs landed while we computed
    __syncthreads();
    cur ^= 1;
  }
#pragma unroll
  for (int j = 0; j < 4; ++j) {
    int nn = nbase + wave * 64 + j * 16 + r;
    float bv = bias[nn];
#pragma unroll
    for (int rr = 0; rr < 8; ++rr) {
      int mm0 = mtile + rr + kh * 8;          // C layout: M = rr + 8*(lane>>4), N = lane&15
      __builtin_nontemporal_store(acc0[j][rr] + bv, &G[(size_t)mm0 * G4 + nn]);
      __builtin_nontemporal_store(acc1[j][rr] + bv, &G[(size_t)(mm0 + 16) * G4 + nn]);
    }
  }
}

// ---------------- 4) fused recurrent step (both directions) ----------------
// grid (4, 64, 2), block 128 (4 waves). K=1024 split 256/wave; A and B frags
// register-pipelined one iteration ahead; LDS reduction; fused gate math.
__global__ void lstm_step(const float* __restrict__ Gf, const float* __restrict__ Gb,
                          const unsigned short* __restrict__ WhTf, const unsigned short* __restrict__ WhTb,
                          const unsigned short* __restrict__ hpf, const unsigned short* __restrict__ hpb,
                          unsigned short* __restrict__ hnf, unsigned short* __restrict__ hnb,
                          float* __restrict__ outp, float* __restrict__ hid, float* __restrict__ cell,
                          int sf, int sb) {
  __shared__ __align__(16) unsigned short Ht[16 * LROW];
  __shared__ __align__(32) float red[4][4][32][8];   // [wave][gate][lane][rr]

  const int dir = blockIdx.z;
  const int s = dir ? sb : sf;
  const float* G = (dir ? Gb : Gf) + (size_t)s * BATCH * G4;
  const unsigned short* WhT = dir ? WhTb : WhTf;
  const unsigned short* hp  = dir ? hpb : hpf;
  unsigned short* hn        = dir ? hnb : hnf;

  const int tid  = threadIdx.x;
  const int lane = tid & 31;
  const int wave = tid >> 5;
  const int mtile = blockIdx.x * 16;                 // over BATCH=64
  const int ntile = blockIdx.y * 16;                 // over H=1024
  const int r = lane & 15, kh = lane >> 4;

  // async-stage h tile (16 x 1024 bf16) once per block
#pragma unroll 4
  for (int row = 0; row < 16; ++row)
    async_copy_b128(Ht + row * LROW + tid * 8,
                    hp + (size_t)(mtile + row) * KDIM + tid * 8);

  const int kbeg = wave * 256;                        // K slice for this wave
  const unsigned short* arow = Ht + r * LROW;
  const unsigned short* brow[4];
#pragma unroll
  for (int g = 0; g < 4; ++g) brow[g] = WhT + (size_t)(g * KDIM + ntile + r) * KDIM;

  // preload first B fragments from global BEFORE waiting on the h staging
  v16bf bcur[4];
#pragma unroll
  for (int g = 0; g < 4; ++g) bcur[g] = load_frag_b_s(brow[g] + kbeg + kh * 16);

  wait_async0();
  __syncthreads();

  v8f acc[4] = {};                                    // i, f, g, o partials
  v16bf acur = load_frag_a(arow, kbeg, kh);
#pragma unroll
  for (int it = 0; it < 8; ++it) {
    const int k0 = kbeg + it * 32;
    v16bf anxt;
    v16bf bnxt[4];
    if (it < 7) {
      anxt = load_frag_a(arow, k0 + 32, kh);
#pragma unroll
      for (int g = 0; g < 4; ++g) bnxt[g] = load_frag_b_s(brow[g] + k0 + 32 + kh * 16);
    }
#pragma unroll
    for (int g = 0; g < 4; ++g)
      acc[g] = __builtin_amdgcn_wmma_f32_16x16x32_bf16(false, acur, false, bcur[g],
                                                       (short)0, acc[g], false, false);
    if (it < 7) {
      acur = anxt;
#pragma unroll
      for (int g = 0; g < 4; ++g) bcur[g] = bnxt[g];
    }
  }
#pragma unroll
  for (int g = 0; g < 4; ++g) *(v8f*)&red[wave][g][lane][0] = acc[g];
  __syncthreads();

  // epilogue: wave handles C-rows rr = 2*wave, 2*wave+1 (all 32 lanes)
  const int nn = ntile + r;
#pragma unroll
  for (int q = 0; q < 2; ++q) {
    const int rr = wave * 2 + q;
    float gi = 0.f, gf = 0.f, gg = 0.f, go = 0.f;
#pragma unroll
    for (int w = 0; w < 4; ++w) {
      gi += red[w][0][lane][rr];
      gf += red[w][1][lane][rr];
      gg += red[w][2][lane][rr];
      go += red[w][3][lane][rr];
    }
    const int mm = mtile + rr + kh * 8;               // batch index
    const size_t gbase = (size_t)mm * G4 + nn;
    gi += __builtin_nontemporal_load(&G[gbase]);           // Gpre is a one-shot stream:
    gf += __builtin_nontemporal_load(&G[gbase + 1024]);    // keep it out of L2 so WhT
    gg += __builtin_nontemporal_load(&G[gbase + 2048]);    // stays resident
    go += __builtin_nontemporal_load(&G[gbase + 3072]);

    const size_t cidx = (size_t)mm * 2048 + dir * 1024 + nn;
    float cn = sigm(gf) * cell[cidx] + sigm(gi) * tanhf(gg);
    float hv = sigm(go) * tanhf(cn);

    cell[cidx] = cn;
    __builtin_nontemporal_store(hv, &outp[((size_t)mm * S_LEN + s) * 2048 + dir * 1024 + nn]);
    hid[cidx] = hv;                                   // final step leaves h_T
    hn[(size_t)mm * KDIM + nn] = f2bf(hv);
  }
}

// ---------------- init: zero h ping-pong buffers and cell state ----------------
__global__ void init_state(unsigned short* h0f, unsigned short* h1f,
                           unsigned short* h0b, unsigned short* h1b,
                           float* __restrict__ cell) {
  int t = blockIdx.x * blockDim.x + threadIdx.x;      // 131072 threads
  cell[t] = 0.0f;
  if (t < BATCH * KDIM) { h0f[t] = 0; h1f[t] = 0; h0b[t] = 0; h1b[t] = 0; }
}

extern "C" void kernel_launch(void* const* d_in, const int* in_sizes, int n_in,
                              void* d_out, int out_size, void* d_ws, size_t ws_size,
                              hipStream_t stream) {
  const int*   inputs = (const int*)  d_in[0];
  const float* tbl    = (const float*)d_in[1];
  const float* Wx_f   = (const float*)d_in[2];
  const float* Wh_f   = (const float*)d_in[3];
  const float* b_f    = (const float*)d_in[4];
  const float* Wx_b   = (const float*)d_in[5];
  const float* Wh_b   = (const float*)d_in[6];
  const float* b_b    = (const float*)d_in[7];

  float* out  = (float*)d_out;                               // (B,S,2H)
  float* hid  = out + (size_t)BATCH * S_LEN * 2048;          // (B,2H)
  float* cell = hid + (size_t)BATCH * 2048;                  // (B,2H)

  char* ws = (char*)d_ws;
  size_t off = 0;
  auto carve = [&](size_t bytes) { char* p = ws + off; off += (bytes + 255) & ~(size_t)255; return p; };

  unsigned short* WxTf = (unsigned short*)carve((size_t)G4 * KDIM * 2);
  unsigned short* WxTb = (unsigned short*)carve((size_t)G4 * KDIM * 2);
  unsigned short* WhTf = (unsigned short*)carve((size_t)G4 * KDIM * 2);
  unsigned short* WhTb = (unsigned short*)carve((size_t)G4 * KDIM * 2);
  unsigned short* X    = (unsigned short*)carve((size_t)S_LEN * BATCH * KDIM * 2);
  float* Gf            = (float*)carve((size_t)S_LEN * BATCH * G4 * 4);
  float* Gb            = (float*)carve((size_t)S_LEN * BATCH * G4 * 4);
  unsigned short* h0f  = (unsigned short*)carve((size_t)BATCH * KDIM * 2);
  unsigned short* h1f  = (unsigned short*)carve((size_t)BATCH * KDIM * 2);
  unsigned short* h0b  = (unsigned short*)carve((size_t)BATCH * KDIM * 2);
  unsigned short* h1b  = (unsigned short*)carve((size_t)BATCH * KDIM * 2);

  transpose_to_bf16<<<dim3(128, 32, 4), dim3(32, 8), 0, stream>>>(
      Wx_f, Wx_b, Wh_f, Wh_b, WxTf, WxTb, WhTf, WhTb);

  embed_gather<<<(S_LEN * BATCH * (KDIM / 8)) / 256, 256, 0, stream>>>(inputs, tbl, X);

  gemm_xw<<<dim3((S_LEN * BATCH) / 32, G4 / 256, 2), 128, 0, stream>>>(
      X, WxTf, WxTb, b_f, b_b, Gf, Gb);

  init_state<<<(BATCH * 2048) / 256, 256, 0, stream>>>(h0f, h1f, h0b, h1b, cell);

  for (int t = 0; t < S_LEN; ++t) {
    const unsigned short* hpf = (t & 1) ? h1f : h0f;
    unsigned short*       hnf = (t & 1) ? h0f : h1f;
    const unsigned short* hpb = (t & 1) ? h1b : h0b;
    unsigned short*       hnb = (t & 1) ? h0b : h1b;
    lstm_step<<<dim3(BATCH / 16, KDIM / 16, 2), 128, 0, stream>>>(
        Gf, Gb, WhTf, WhTb, hpf, hpb, hnf, hnb, out, hid, cell, t, S_LEN - 1 - t);
  }
}
